// BiLSTM_CRF_21277267984677
// MI455X (gfx1250) — matrix operand order repaired and checked by reading, not compiled, not run on hardware
//
#include <hip/hip_runtime.h>
#include <hip/hip_bf16.h>
#include <math.h>

#define B_ 128
#define T_ 256
#define E_ 256
#define H_ 512
#define H2_ 256
#define G_ 1024            // 4*H2
#define NTAGS 76
#define NTP 80             // padded tag count
#define START_IDX 74
#define STOP_IDX 75
#define NEG_ -10000.0f

typedef __attribute__((ext_vector_type(16))) _Float16 v16h;
typedef __attribute__((ext_vector_type(8)))  _Float16 v8h;
typedef __attribute__((ext_vector_type(8)))  float    v8f;

// ---- WMMA fragment loaders (16x16x32 f16, wave32) -------------------------
// A (16x32): lanes0-15 m=L, K: e<8 -> k0+e ; e>=8 -> k0+16+(e-8); lanes16-31 +8
__device__ __forceinline__ v16h load_fragA(const _Float16* A, int lda, int row,
                                           int k0, int laneHi) {
  const _Float16* p = A + (size_t)row * lda + k0 + 8 * laneHi;
  v8h lo = *(const v8h*)(p);
  v8h hi = *(const v8h*)(p + 16);
  v16h r;
#pragma unroll
  for (int i = 0; i < 8; ++i) { r[i] = lo[i]; r[i + 8] = hi[i]; }
  return r;
}
// B (32x16): lane n=L&15; K: e<8 -> k0+e ; e>=8 -> k0+16+(e-8); lanes16-31 +8
__device__ __forceinline__ v16h load_fragB(const _Float16* Bm, int ldb, int nrow,
                                           int k0, int laneHi) {
  const _Float16* p = Bm + (size_t)nrow * ldb + k0 + 8 * laneHi;
  v8h lo = *(const v8h*)(p);
  v8h hi = *(const v8h*)(p + 16);
  v16h r;
#pragma unroll
  for (int i = 0; i < 8; ++i) { r[i] = lo[i]; r[i + 8] = hi[i]; }
  return r;
}

__device__ __forceinline__ float sigm(float x) { return 1.f / (1.f + __expf(-x)); }

// ---- prep: f32->f16 weight conversion, bias sums, padded W_out, h0/c0 -----
__global__ void prep_kernel(const float* __restrict__ Wih_f, const float* __restrict__ Whh_f,
                            const float* __restrict__ bih_f, const float* __restrict__ bhh_f,
                            const float* __restrict__ Wih_b, const float* __restrict__ Whh_b,
                            const float* __restrict__ bih_b, const float* __restrict__ bhh_b,
                            const float* __restrict__ h0,    const float* __restrict__ c0,
                            const float* __restrict__ W_out, const float* __restrict__ b_out,
                            _Float16* __restrict__ Wih16, _Float16* __restrict__ Whh16,
                            _Float16* __restrict__ Wout16, float* __restrict__ bias_sum,
                            float* __restrict__ bout_pad, _Float16* __restrict__ h16,
                            float* __restrict__ cbuf) {
  int tid = blockIdx.x * blockDim.x + threadIdx.x;
  const int WSZ = G_ * E_;                       // 262144 (E_ == H2_)
  if (tid < WSZ) { Wih16[tid] = (_Float16)Wih_f[tid]; return; }
  tid -= WSZ;
  if (tid < WSZ) { Wih16[WSZ + tid] = (_Float16)Wih_b[tid]; return; }
  tid -= WSZ;
  if (tid < WSZ) { Whh16[tid] = (_Float16)Whh_f[tid]; return; }
  tid -= WSZ;
  if (tid < WSZ) { Whh16[WSZ + tid] = (_Float16)Whh_b[tid]; return; }
  tid -= WSZ;
  if (tid < NTP * H_) {
    int n = tid / H_, k = tid % H_;
    Wout16[tid] = (n < NTAGS) ? (_Float16)W_out[n * H_ + k] : (_Float16)0.f;
    return;
  }
  tid -= NTP * H_;
  if (tid < 2 * G_) {
    bias_sum[tid] = (tid < G_) ? (bih_f[tid] + bhh_f[tid])
                               : (bih_b[tid - G_] + bhh_b[tid - G_]);
    return;
  }
  tid -= 2 * G_;
  if (tid < NTP) { bout_pad[tid] = (tid < NTAGS) ? b_out[tid] : 0.f; return; }
  tid -= NTP;
  if (tid < 2 * B_ * H2_) { h16[tid] = (_Float16)h0[tid]; return; }
  tid -= 2 * B_ * H2_;
  if (tid < 2 * B_ * H2_) { cbuf[tid] = c0[tid]; return; }
}

// ---- embedding gather -> f16 ----------------------------------------------
__global__ void embed_kernel(const int* __restrict__ ids, const float* __restrict__ tbl,
                             _Float16* __restrict__ emb16) {
  int tid = blockIdx.x * blockDim.x + threadIdx.x;   // B*T*64 threads, 4 elems each
  int row = tid >> 6;
  int e = (tid & 63) << 2;
  int id = ids[row];
  const float* s = tbl + (size_t)id * E_ + e;
  _Float16* d = emb16 + (size_t)row * E_ + e;
  d[0] = (_Float16)s[0]; d[1] = (_Float16)s[1];
  d[2] = (_Float16)s[2]; d[3] = (_Float16)s[3];
}

// ---- big time-batched input projection: xW = emb @ Wih^T + (bih+bhh) ------
// 65536 waves: dir(2) x Mtile(2048) x Njob(16 x 64cols)
__global__ void xw_gemm_kernel(const _Float16* __restrict__ A,
                               const _Float16* __restrict__ W,
                               const float* __restrict__ bias,
                               float* __restrict__ xW) {
  int wid = (blockIdx.x * blockDim.x + threadIdx.x) >> 5;
  int lane = threadIdx.x & 31;
  int laneHi = lane >> 4, lanelo = lane & 15;
  int dir = wid >> 15;
  int rem = wid & 32767;
  int m0 = (rem >> 4) * 16;
  int n0 = (rem & 15) * 64;
  const _Float16* Bm = W + dir * (G_ * E_);
  const float* bs = bias + dir * G_;
  v8f acc[4];
#pragma unroll
  for (int j = 0; j < 4; ++j) {
    float bv = bs[n0 + 16 * j + lanelo];
#pragma unroll
    for (int r = 0; r < 8; ++r) acc[j][r] = bv;
  }
  for (int k0 = 0; k0 < E_; k0 += 32) {
    v16h a = load_fragA(A, E_, m0 + lanelo, k0, laneHi);
#pragma unroll
    for (int j = 0; j < 4; ++j) {
      v16h b = load_fragB(Bm, E_, n0 + 16 * j + lanelo, k0, laneHi);
      acc[j] = __builtin_amdgcn_wmma_f32_16x16x32_f16(false, a, false, b,
                                                      (short)0, acc[j], false, false);
    }
  }
  float* C = xW + (size_t)dir * 32768 * G_;
#pragma unroll
  for (int j = 0; j < 4; ++j) {
    int n = n0 + 16 * j + lanelo;
#pragma unroll
    for (int r = 0; r < 8; ++r)
      C[(size_t)(m0 + r + 8 * laneHi) * G_ + n] = acc[j][r];
  }
}

// ---- fused LSTM step: gates = h@Whh^T + xW[t]; apply activations ----------
// 256 waves: dir(2) x mtile(8) x hidden-col-tile(16). Each wave owns a
// 16(batch)x16(hidden) cell tile and computes its 4 gate tiles (i,f,g,o)
// as 4 accumulators -> elementwise LSTM update entirely in-register.
// h double-buffered (hin read, hout written) to avoid cross-block races.
__global__ void lstm_step_kernel(const _Float16* __restrict__ hin,
                                 _Float16* __restrict__ hout,
                                 const _Float16* __restrict__ Whh,
                                 const float* __restrict__ xW,
                                 float* __restrict__ cbuf,
                                 _Float16* __restrict__ lstm16, int t) {
  int wid = (blockIdx.x * blockDim.x + threadIdx.x) >> 5;   // 0..255
  int lane = threadIdx.x & 31;
  int laneHi = lane >> 4, lanelo = lane & 15;
  int dir = wid >> 7;
  int rem = wid & 127;
  int m0 = (rem >> 4) * 16;          // batch tile
  int j0 = (rem & 15) * 16;          // hidden-unit tile
  int tt = dir ? (T_ - 1 - t) : t;
  const _Float16* A = hin + dir * (B_ * H2_);
  const _Float16* Bm = Whh + dir * (G_ * H2_);
  const float* Cin = xW + (size_t)dir * 32768 * G_;
  v8f acc[4];                         // gate g at columns g*256 + j0..j0+15
#pragma unroll
  for (int g = 0; g < 4; ++g) {
    int n = g * H2_ + j0 + lanelo;
#pragma unroll
    for (int r = 0; r < 8; ++r)
      acc[g][r] = Cin[((size_t)(m0 + r + 8 * laneHi) * T_ + tt) * G_ + n];
  }
  for (int k0 = 0; k0 < H2_; k0 += 32) {
    v16h a = load_fragA(A, H2_, m0 + lanelo, k0, laneHi);
#pragma unroll
    for (int g = 0; g < 4; ++g) {
      v16h b = load_fragB(Bm, H2_, g * H2_ + j0 + lanelo, k0, laneHi);
      acc[g] = __builtin_amdgcn_wmma_f32_16x16x32_f16(false, a, false, b,
                                                      (short)0, acc[g], false, false);
    }
  }
  // activations: lane holds (m = m0+r+8*laneHi, hidden col = j0+lanelo)
  int tpos = dir ? (T_ - 1 - t) : t;
  int hcol = j0 + lanelo;
  int colOut = dir ? (H2_ + hcol) : hcol;
#pragma unroll
  for (int r = 0; r < 8; ++r) {
    int m = m0 + r + 8 * laneHi;
    size_t sidx = (size_t)dir * (B_ * H2_) + (size_t)m * H2_ + hcol;
    float c = cbuf[sidx];
    float si = sigm(acc[0][r]);
    float sf = sigm(acc[1][r]);
    float gg = tanhf(acc[2][r]);
    float so = sigm(acc[3][r]);
    c = sf * c + si * gg;
    float h = so * tanhf(c);
    cbuf[sidx] = c;
    hout[sidx] = (_Float16)h;
    lstm16[((size_t)m * T_ + tpos) * H_ + colOut] = (_Float16)h;
  }
}

// ---- feats = lstm_out @ W_out^T + b_out (N padded to 80) ------------------
__global__ void feats_gemm_kernel(const _Float16* __restrict__ A,
                                  const _Float16* __restrict__ Wout,
                                  const float* __restrict__ bout,
                                  float* __restrict__ feats) {
  int wid = (blockIdx.x * blockDim.x + threadIdx.x) >> 5;   // 0..2047
  int lane = threadIdx.x & 31;
  int laneHi = lane >> 4, lanelo = lane & 15;
  int m0 = wid * 16;
  v8f acc[5];
#pragma unroll
  for (int j = 0; j < 5; ++j) {
    float bv = bout[16 * j + lanelo];
#pragma unroll
    for (int r = 0; r < 8; ++r) acc[j][r] = bv;
  }
  for (int k0 = 0; k0 < H_; k0 += 32) {
    v16h a = load_fragA(A, H_, m0 + lanelo, k0, laneHi);
#pragma unroll
    for (int j = 0; j < 5; ++j) {
      v16h b = load_fragB(Wout, H_, 16 * j + lanelo, k0, laneHi);
      acc[j] = __builtin_amdgcn_wmma_f32_16x16x32_f16(false, a, false, b,
                                                      (short)0, acc[j], false, false);
    }
  }
#pragma unroll
  for (int j = 0; j < 5; ++j) {
    int n = 16 * j + lanelo;
#pragma unroll
    for (int r = 0; r < 8; ++r)
      feats[(size_t)(m0 + r + 8 * laneHi) * NTP + n] = acc[j][r];
  }
}

// ---- Viterbi: one block per batch row; transitions cached in LDS ----------
__global__ void viterbi_kernel(const float* __restrict__ feats,
                               const float* __restrict__ trans,
                               unsigned char* __restrict__ bptr,
                               float* __restrict__ out) {
  __shared__ float st[NTAGS * NTAGS];
  __shared__ float fv[NTAGS];
  __shared__ float fvn[NTAGS];
  int b = blockIdx.x;
  int n = threadIdx.x;
  for (int i = n; i < NTAGS * NTAGS; i += blockDim.x) st[i] = trans[i];
  if (n < NTAGS) fv[n] = (n == START_IDX) ? 0.f : NEG_;
  __syncthreads();
  for (int t = 0; t < T_; ++t) {
    if (n < NTAGS) {
      float best = -3.4e38f; int bp = 0;
      const float* tr = st + n * NTAGS;
      for (int p = 0; p < NTAGS; ++p) {
        float s = fv[p] + tr[p];
        if (s > best) { best = s; bp = p; }       // first-max, matches argmax
      }
      fvn[n] = best + feats[((size_t)b * T_ + t) * NTP + n];
      bptr[((size_t)b * T_ + t) * NTAGS + n] = (unsigned char)bp;
    }
    __syncthreads();
    if (n < NTAGS) fv[n] = fvn[n];
    __syncthreads();
  }
  if (n == 0) {
    float best = -3.4e38f; int bt = 0;
    for (int p = 0; p < NTAGS; ++p) {
      float s = fv[p] + st[STOP_IDX * NTAGS + p];
      if (s > best) { best = s; bt = p; }
    }
    out[b] = best;
    int tag = bt;
    for (int t = T_ - 1; t >= 0; --t) {
      out[B_ + b * T_ + t] = (float)tag;
      tag = bptr[((size_t)b * T_ + t) * NTAGS + tag];
    }
  }
}

extern "C" void kernel_launch(void* const* d_in, const int* in_sizes, int n_in,
                              void* d_out, int out_size, void* d_ws, size_t ws_size,
                              hipStream_t stream) {
  (void)in_sizes; (void)n_in; (void)out_size; (void)ws_size;
  const int*   input_ids   = (const int*)d_in[0];
  const float* embed_table = (const float*)d_in[1];
  const float* Wih_f = (const float*)d_in[2];
  const float* Whh_f = (const float*)d_in[3];
  const float* bih_f = (const float*)d_in[4];
  const float* bhh_f = (const float*)d_in[5];
  const float* Wih_b = (const float*)d_in[6];
  const float* Whh_b = (const float*)d_in[7];
  const float* bih_b = (const float*)d_in[8];
  const float* bhh_b = (const float*)d_in[9];
  const float* h0    = (const float*)d_in[10];
  const float* c0    = (const float*)d_in[11];
  const float* W_out = (const float*)d_in[12];
  const float* b_out = (const float*)d_in[13];
  const float* transitions = (const float*)d_in[14];
  float* out = (float*)d_out;

  char* ws = (char*)d_ws;
  size_t off = 0;
  auto carve = [&](size_t bytes) -> char* {
    char* p = ws + off;
    off += (bytes + 255) & ~(size_t)255;
    return p;
  };
  const size_t ROWS = (size_t)B_ * T_;                       // 32768
  _Float16* emb16   = (_Float16*)carve(ROWS * E_ * 2);       // 16 MB
  _Float16* Wih16   = (_Float16*)carve(2u * G_ * E_ * 2);    // 1 MB
  _Float16* Whh16   = (_Float16*)carve(2u * G_ * H2_ * 2);   // 1 MB
  _Float16* Wout16  = (_Float16*)carve((size_t)NTP * H_ * 2);
  float*    biasSum = (float*)carve(2u * G_ * 4);
  float*    boutPad = (float*)carve((size_t)NTP * 4);
  float*    xW      = (float*)carve(2 * ROWS * G_ * 4);      // 268 MB
  _Float16* h16a    = (_Float16*)carve(2u * B_ * H2_ * 2);   // ping
  _Float16* h16b    = (_Float16*)carve(2u * B_ * H2_ * 2);   // pong
  float*    cbuf    = (float*)carve(2u * B_ * H2_ * 4);
  _Float16* lstm16  = (_Float16*)carve(ROWS * H_ * 2);       // 33 MB
  float*    feats   = (float*)carve(ROWS * NTP * 4);         // 10 MB
  unsigned char* bptr = (unsigned char*)carve(ROWS * NTAGS); // 2.5 MB

  // 1) prep
  {
    int total = 4 * (G_ * E_) + NTP * H_ + 2 * G_ + NTP + 2 * (2 * B_ * H2_);
    prep_kernel<<<(total + 255) / 256, 256, 0, stream>>>(
        Wih_f, Whh_f, bih_f, bhh_f, Wih_b, Whh_b, bih_b, bhh_b,
        h0, c0, W_out, b_out,
        Wih16, Whh16, Wout16, biasSum, boutPad, h16a, cbuf);
  }
  // 2) embedding gather -> f16
  embed_kernel<<<(int)(ROWS * 64 / 256), 256, 0, stream>>>(input_ids, embed_table, emb16);
  // 3) time-batched input projection (both directions)
  xw_gemm_kernel<<<8192, 256, 0, stream>>>(emb16, Wih16, biasSum, xW);
  // 4) recurrence: 256 fused gate-GEMM + activation steps, h ping-pong
  for (int t = 0; t < T_; ++t) {
    const _Float16* hin = (t & 1) ? h16b : h16a;
    _Float16* hout      = (t & 1) ? h16a : h16b;
    lstm_step_kernel<<<32, 256, 0, stream>>>(hin, hout, Whh16, xW, cbuf, lstm16, t);
  }
  // 5) output projection
  feats_gemm_kernel<<<256, 256, 0, stream>>>(lstm16, Wout16, boutPad, feats);
  // 6) Viterbi decode + backtrace
  viterbi_kernel<<<B_, 128, 0, stream>>>(feats, transitions, bptr, out);
}
